// IRNN_second_61297773248930
// MI455X (gfx1250) — matrix operand order repaired and checked by reading, not compiled, not run on hardware
//
#include <hip/hip_runtime.h>
#include <hip/hip_bf16.h>

// Problem sizes (fixed by the reference)
#define B_ 128
#define T_ 1024
#define I_ 256
#define H_ 1024
#define O_ 256

typedef __attribute__((ext_vector_type(16))) __bf16 v16bf;
typedef __attribute__((ext_vector_type(8)))  __bf16 v8bf;
typedef __attribute__((ext_vector_type(8)))  float  v8f;

// Workspace layout (bytes)
#define WHH_OFF  0
#define WHH_BYTES ((size_t)H_ * H_ * 2)                    // 2 MB  bf16 W_hh [H][H]
#define WIH_OFF  (WHH_OFF + WHH_BYTES)
#define WIH_BYTES ((size_t)H_ * I_ * 2)                    // 0.5 MB bf16 W_ih [H][I]
#define INPBF_OFF (WIH_OFF + WIH_BYTES)
#define INPBF_BYTES ((size_t)B_ * T_ * I_ * 2)             // 64 MB bf16 inp [B][T][I]
#define HFIN_OFF (INPBF_OFF + INPBF_BYTES)
#define HFIN_BYTES ((size_t)B_ * H_ * 4)                   // 0.5 MB f32 h_final [B][H]

// ---------------------------------------------------------------- cvt f32->bf16
__global__ void cvt_f32_to_bf16(const float* __restrict__ src,
                                __bf16* __restrict__ dst, int n) {
  int i = blockIdx.x * blockDim.x + threadIdx.x;
  if (i < n) dst[i] = (__bf16)src[i];
}

// ---------------------------------------------------------------- fused scan
// Grid: 8 workgroups (one 16-row batch tile each), 512 threads = 16 waves.
// Each wave owns 64 columns of h (4 WMMA N-tiles). Per step, per wave:
// 40 k-chunks x 4 tiles = 160 v_wmma_f32_16x16x32_bf16, with the B-operand
// stream (W_hh / W_ih from L2) software-pipelined one full k-chunk ahead.
__global__ __launch_bounds__(512)
void irnn_scan(const __bf16* __restrict__ inp_bf,   // [B][T][I] bf16
               const float*  __restrict__ hprev,    // [B][H] f32
               const __bf16* __restrict__ wih,      // [H][I] bf16
               const __bf16* __restrict__ whh,      // [H][H] bf16
               const float*  __restrict__ b_ih,     // [H]
               const float*  __restrict__ b_hh,     // [H]
               float*        __restrict__ hfinal)   // [B][H] f32
{
  __shared__ __bf16 hbuf[2][16][H_];                // 64 KB double-buffered h tile

  const int tid   = threadIdx.x;
  const int lane  = tid & 31;
  const int wave  = tid >> 5;
  const int b0    = blockIdx.x * 16;
  const int lhalf = lane & 15;
  const bool hi   = lane >= 16;

  // init h tile from hPrevious
  for (int idx = tid; idx < 16 * H_; idx += 512) {
    int m = idx >> 10, k = idx & (H_ - 1);
    hbuf[0][m][k] = (__bf16)hprev[(size_t)(b0 + m) * H_ + k];
  }
  __syncthreads();

  const int nbase = wave * 64;                      // 4 n-tiles: nbase .. nbase+63
  float bias[4];
#pragma unroll
  for (int j = 0; j < 4; ++j) {
    int col = nbase + j * 16 + lhalf;
    bias[j] = b_ih[col] + b_hh[col];
  }

  // A-operand (16x32 bf16, contiguous K to match B): lane<16 row=lane, K c..c+15;
  // lane>=16 row=lane-16, K c+16..c+31
  const int am  = lhalf;
  const int ak0 = hi ? 16 : 0;
  const int ak1 = ak0 + 8;
  // B-operand (32x16 bf16): lane<16 col=lane, K 0..15; lane>=16 col, K 16..31
  const int bk  = hi ? 16 : 0;

  // Per-tile weight row base pointers (loop-invariant)
  const __bf16* whh_base[4];
  const __bf16* wih_base[4];
#pragma unroll
  for (int j = 0; j < 4; ++j) {
    whh_base[j] = whh + (size_t)(nbase + j * 16 + lhalf) * H_ + bk;
    wih_base[j] = wih + (size_t)(nbase + j * 16 + lhalf) * I_ + bk;
  }

  int p = 0;
  for (int t = 0; t < T_; ++t) {
    v8f acc[4];
#pragma unroll
    for (int j = 0; j < 4; ++j) acc[j] = (v8f){};

    // Double-buffered B-operand registers: one full k-chunk of prefetch distance
    v16bf bbuf[2][4];
#pragma unroll
    for (int j = 0; j < 4; ++j) bbuf[0][j] = *(const v16bf*)(whh_base[j]);

    // ---- recurrent part of the reduction: K = H over LDS-resident h ----
#pragma unroll 2
    for (int kc = 0; kc < H_ / 32; ++kc) {
      const int cur = kc & 1, nxt = cur ^ 1;
      const int c = kc * 32;
      v8bf alo = *(const v8bf*)&hbuf[p][am][c + ak0];
      v8bf ahi = *(const v8bf*)&hbuf[p][am][c + ak1];
      v16bf a = __builtin_shufflevector(alo, ahi,
                 0, 1, 2, 3, 4, 5, 6, 7, 8, 9, 10, 11, 12, 13, 14, 15);
      // preload next chunk's B while WMMAs below run on the current one
      if (kc < H_ / 32 - 1) {
#pragma unroll
        for (int j = 0; j < 4; ++j)
          bbuf[nxt][j] = *(const v16bf*)(whh_base[j] + c + 32);
      } else {
#pragma unroll
        for (int j = 0; j < 4; ++j)        // carry pipeline into the W_ih phase
          bbuf[nxt][j] = *(const v16bf*)(wih_base[j]);
      }
#pragma unroll
      for (int j = 0; j < 4; ++j)
        acc[j] = __builtin_amdgcn_wmma_f32_16x16x32_bf16(
                     false, a, false, bbuf[cur][j], (short)0, acc[j], false, false);
    }

    // ---- fused input projection: K = I over inp[b0..b0+15][t][:] ----
    // (H_/32 is even, so W_ih chunk 0 sits in bbuf[0] -> ic parity lines up)
    const __bf16* ap_row = inp_bf + ((size_t)(b0 + am) * T_ + t) * I_;
#pragma unroll 2
    for (int ic = 0; ic < I_ / 32; ++ic) {
      const int cur = ic & 1, nxt = cur ^ 1;
      const int c = ic * 32;
      v8bf alo = *(const v8bf*)(ap_row + c + ak0);
      v8bf ahi = *(const v8bf*)(ap_row + c + ak1);
      v16bf a = __builtin_shufflevector(alo, ahi,
                 0, 1, 2, 3, 4, 5, 6, 7, 8, 9, 10, 11, 12, 13, 14, 15);
      if (ic < I_ / 32 - 1) {
#pragma unroll
        for (int j = 0; j < 4; ++j)
          bbuf[nxt][j] = *(const v16bf*)(wih_base[j] + c + 32);
      }
#pragma unroll
      for (int j = 0; j < 4; ++j)
        acc[j] = __builtin_amdgcn_wmma_f32_16x16x32_bf16(
                     false, a, false, bbuf[cur][j], (short)0, acc[j], false, false);
    }

    // ---- epilogue: bias + relu, write bf16 h into the other LDS buffer ----
    // C/D layout: VGPR r -> row r (lanes 0-15) / row r+8 (lanes 16-31)
    const int q = p ^ 1;
    const int rbase = hi ? 8 : 0;
#pragma unroll
    for (int j = 0; j < 4; ++j) {
      const int col = nbase + j * 16 + lhalf;
#pragma unroll
      for (int r = 0; r < 8; ++r) {
        float v = acc[j][r] + bias[j];
        v = v > 0.0f ? v : 0.0f;
        hbuf[q][rbase + r][col] = (__bf16)v;
      }
    }
    // hint: stream next timestep's input slice toward the caches
    if (t + 1 < T_)
      __builtin_prefetch(ap_row + I_, 0, 0);
    __syncthreads();
    p = q;
  }

  // write final hidden state (f32) for the output projection
  for (int idx = tid; idx < 16 * H_; idx += 512) {
    int m = idx >> 10, k = idx & (H_ - 1);
    hfinal[(size_t)(b0 + m) * H_ + k] = (float)hbuf[p][m][k];
  }
}

// ---------------------------------------------------------------- out = h @ W_out^T + b
__global__ __launch_bounds__(256)
void out_proj(const float* __restrict__ hfinal,     // [B][H]
              const float* __restrict__ wout,       // [O][H]
              const float* __restrict__ bout,       // [O]
              float* __restrict__ out) {            // [B][O]
  const int b = blockIdx.x;
  const int o = threadIdx.x;
  const float* hr = hfinal + (size_t)b * H_;
  const float* wr = wout + (size_t)o * H_;
  float acc = bout[o];
#pragma unroll 8
  for (int k = 0; k < H_; ++k) acc = fmaf(hr[k], wr[k], acc);
  out[(size_t)b * O_ + o] = acc;
}

// ---------------------------------------------------------------- launch
extern "C" void kernel_launch(void* const* d_in, const int* in_sizes, int n_in,
                              void* d_out, int out_size, void* d_ws, size_t ws_size,
                              hipStream_t stream) {
  const float* inp   = (const float*)d_in[0];   // [B,T,I]
  const float* hprev = (const float*)d_in[1];   // [1,B,H]
  const float* wih   = (const float*)d_in[2];   // [H,I]
  const float* whh   = (const float*)d_in[3];   // [H,H]
  const float* bih   = (const float*)d_in[4];   // [H]
  const float* bhh   = (const float*)d_in[5];   // [H]
  const float* wout  = (const float*)d_in[6];   // [O,H]
  const float* bout  = (const float*)d_in[7];   // [O]
  float* out = (float*)d_out;

  char* ws = (char*)d_ws;
  __bf16* whh_bf = (__bf16*)(ws + WHH_OFF);
  __bf16* wih_bf = (__bf16*)(ws + WIH_OFF);
  __bf16* inp_bf = (__bf16*)(ws + INPBF_OFF);
  float*  hfin   = (float*)(ws + HFIN_OFF);

  cvt_f32_to_bf16<<<(H_ * H_ + 255) / 256, 256, 0, stream>>>(whh, whh_bf, H_ * H_);
  cvt_f32_to_bf16<<<(H_ * I_ + 255) / 256, 256, 0, stream>>>(wih, wih_bf, H_ * I_);
  cvt_f32_to_bf16<<<(B_ * T_ * I_ + 255) / 256, 256, 0, stream>>>(inp, inp_bf, B_ * T_ * I_);

  irnn_scan<<<B_ / 16, 512, 0, stream>>>(inp_bf, hprev, wih_bf, whh_bf, bih, bhh, hfin);

  out_proj<<<B_, O_, 0, stream>>>(hfin, wout, bout, out);
}
